// model_alexnet_v0_75153337745893
// MI455X (gfx1250) — matrix operand order repaired and checked by reading, not compiled, not run on hardware
//
#include <hip/hip_runtime.h>
#include <hip/hip_bf16.h>

typedef __attribute__((ext_vector_type(16))) _Float16 v16h;
typedef __attribute__((ext_vector_type(8)))  _Float16 v8h;
typedef __attribute__((ext_vector_type(8)))  float    v8f;

// ---------------------------------------------------------------- async copy helpers (CDNA5)
// global -> LDS async copy, 16 bytes, tracked by ASYNCcnt.
__device__ __forceinline__ void async_b128_to_lds(void* ldsp, const void* gaddr) {
  uint32_t ldsoff = (uint32_t)(uintptr_t)ldsp;   // flat shared addr: low 32 bits = LDS offset
  asm volatile("global_load_async_to_lds_b128 %0, %1, off"
               :: "v"(ldsoff), "v"(gaddr) : "memory");
}
__device__ __forceinline__ void wait_async0() {
  asm volatile("s_wait_asynccnt 0x0" ::: "memory");
}

// ---------------------------------------------------------------- conversions
__global__ void k_f32_to_f16(const float* __restrict__ in, _Float16* __restrict__ out, int n) {
  int i = blockIdx.x * blockDim.x + threadIdx.x;
  if (i < n) out[i] = (_Float16)in[i];
}

__global__ void k_w_pad_f16(const float* __restrict__ in, _Float16* __restrict__ out,
                            int O, int K, int Kpad) {
  int i = blockIdx.x * blockDim.x + threadIdx.x;
  if (i >= O * Kpad) return;
  int o = i / Kpad, k = i - o * Kpad;
  out[i] = (k < K) ? (_Float16)in[o * K + k] : (_Float16)0.0f;
}

// ---------------------------------------------------------------- maxpool 3x3 s2 (valid)
__global__ void k_maxpool3s2(const _Float16* __restrict__ in, _Float16* __restrict__ out,
                             int C, int H, int W, int PH, int PW, int total) {
  int i = blockIdx.x * blockDim.x + threadIdx.x;
  if (i >= total) return;
  int pw = i % PW; int t = i / PW;
  int ph = t % PH; t /= PH;
  int c  = t % C;  int b = t / C;
  const _Float16* p = in + (((size_t)b * C + c) * H + ph * 2) * W + pw * 2;
  float m = -3.4e38f;
  #pragma unroll
  for (int dy = 0; dy < 3; ++dy)
    #pragma unroll
    for (int dx = 0; dx < 3; ++dx) {
      float v = (float)p[dy * W + dx];
      m = v > m ? v : m;
    }
  out[i] = (_Float16)m;
}

// ---------------------------------------------------------------- implicit-GEMM conv, WMMA f16->f32
// Block: 8 waves, tile 128(M) x 64(N). Weight tile (64 cols x 32 K halves = 4KB) staged in LDS
// via async global->LDS, double buffered; all 8 waves read B fragments from LDS (8x traffic cut).
// A frag (16-bit 16x32, ISA 7.12.2): lane row M=lane&15, kb=(lane>=16)*16, elem e -> K=k0+kb+e (this code's contiguous order, same as B).
// B frag: lane col N=lane&15, K = (lane>=16?16:0)+e -> contiguous 32B of the staged column.
template<int KH, int KW, int STRIDE, int PAD>
__global__ __launch_bounds__(256) void k_conv_wmma(
    const _Float16* __restrict__ in, const _Float16* __restrict__ wt,
    const float* __restrict__ bias, _Float16* __restrict__ out,
    int Bn, int C, int H, int W, int O, int OH, int OW, int K, int Kpad) {
  __shared__ _Float16 ldsB[2][64 * 32];

  const int tid   = threadIdx.x;
  const int lane  = tid & 31;
  const int wid   = tid >> 5;
  const int mtile = blockIdx.x * 8 + wid;
  const int nbase = blockIdx.y * 64;
  const int l15   = lane & 15;
  const int hi    = lane >> 4;
  const int kbA   = hi * 8;
  const int kbB   = hi * 16;
  constexpr int KHKW = KH * KW;

  // loader role: each thread owns one 16B chunk of the 4KB tile
  const int lcol   = tid >> 2;       // 0..63
  const int lchunk = tid & 3;        // 0..3  (8 halves each)
  const _Float16* gB = wt + (size_t)(nbase + lcol) * Kpad + lchunk * 8;
  _Float16* sB0 = &ldsB[0][lcol * 32 + lchunk * 8];
  _Float16* sB1 = &ldsB[1][lcol * 32 + lchunk * 8];

  const int M   = Bn * OH * OW;
  int row = mtile * 16 + l15;
  if (row >= M) row = M - 1;         // clamp (no early return: barriers below)
  const int ow  = row % OW;
  int t0 = row / OW;
  const int oh  = t0 % OH;
  const int b   = t0 / OH;

  v8f acc[4] = {};

  // prologue: stage K-step 0 into buffer 0
  async_b128_to_lds(sB0, gB);
  wait_async0();
  __syncthreads();

  const int steps = Kpad >> 5;
  for (int s = 0; s < steps; ++s) {
    const int k0  = s << 5;
    const int buf = s & 1;
    if (s + 1 < steps)                         // overlap next tile load with compute
      async_b128_to_lds(buf ? sB0 : sB1, gB + (k0 + 32));

    // A fragment: on-the-fly im2col gather (compile-time KH/KW -> constant div)
    v16h a = {};
    #pragma unroll
    for (int e = 0; e < 16; ++e) {
      int k = k0 + kbB + e;
      _Float16 val = (_Float16)0.0f;
      if (k < K) {
        int c  = k / KHKW;
        int r  = k - c * KHKW;
        int kh = r / KW;
        int kw = r - kh * KW;
        int ih = oh * STRIDE - PAD + kh;
        int iw = ow * STRIDE - PAD + kw;
        if (ih >= 0 && ih < H && iw >= 0 && iw < W)
          val = in[(((size_t)b * C + c) * H + ih) * W + iw];
      }
      a[e] = val;
    }

    #pragma unroll
    for (int nt = 0; nt < 4; ++nt) {
      v16h bf = *(const v16h*)(&ldsB[buf][(nt * 16 + l15) * 32 + kbB]);
      acc[nt] = __builtin_amdgcn_wmma_f32_16x16x32_f16(false, a, false, bf,
                                                       (short)0, acc[nt], false, false);
    }

    wait_async0();          // next buffer landed (own issue; barrier covers the rest)
    __syncthreads();        // all loaders done + all readers done with this buffer
  }

  #pragma unroll
  for (int r = 0; r < 8; ++r) {
    int mr = mtile * 16 + r + hi * 8;          // C/D layout: vgpr r -> M=r / M=8+r
    if (mr >= M) continue;
    int oww = mr % OW;
    int tt  = mr / OW;
    int ohh = tt % OH;
    int bb  = tt / OH;
    #pragma unroll
    for (int nt = 0; nt < 4; ++nt) {
      int n = nbase + nt * 16 + l15;
      float v = acc[nt][r] + bias[n];
      v = v > 0.f ? v : 0.f;
      out[(((size_t)bb * O + n) * OH + ohh) * OW + oww] = (_Float16)v;
    }
  }
}

// ---------------------------------------------------------------- dense GEMM: out = act(A[M,K] x Bt[N,K]^T + bias)
// Same LDS-staged B pipeline; A rows loaded directly (b128, no cross-wave reuse).
__global__ __launch_bounds__(256) void k_gemm_wmma(
    const _Float16* __restrict__ A, const _Float16* __restrict__ Bt,
    const float* __restrict__ bias, void* __restrict__ outp,
    int M, int N, int K, int ldc, int act, int out_f32) {
  __shared__ _Float16 ldsB[2][64 * 32];

  const int tid   = threadIdx.x;
  const int lane  = tid & 31;
  const int wid   = tid >> 5;
  const int mtile = blockIdx.x * 8 + wid;
  const int nbase = blockIdx.y * 64;
  const int l15   = lane & 15;
  const int hi    = lane >> 4;
  const int kbA   = hi * 8;
  const int kbB   = hi * 16;

  const int lcol   = tid >> 2;
  const int lchunk = tid & 3;
  const bool colOK = (nbase + lcol) < N;
  const _Float16* gB = Bt + (size_t)(nbase + lcol) * K + lchunk * 8;
  _Float16* sB0 = &ldsB[0][lcol * 32 + lchunk * 8];
  _Float16* sB1 = &ldsB[1][lcol * 32 + lchunk * 8];
  if (!colOK) {                       // zero the out-of-range columns once (both buffers)
    v8h z = {};
    *(v8h*)sB0 = z;
    *(v8h*)sB1 = z;
  }

  int row = mtile * 16 + l15;
  if (row >= M) row = M - 1;
  const _Float16* Ar = A + (size_t)row * K;

  v8f acc[4] = {};

  if (colOK) async_b128_to_lds(sB0, gB);
  wait_async0();
  __syncthreads();

  const int steps = K >> 5;
  for (int s = 0; s < steps; ++s) {
    const int k0  = s << 5;
    const int buf = s & 1;
    if (colOK && (s + 1 < steps))
      async_b128_to_lds(buf ? sB0 : sB1, gB + (k0 + 32));

    v8h lo = *(const v8h*)(Ar + k0 + kbB);
    v8h hb = *(const v8h*)(Ar + k0 + kbB + 8);
    v16h a = __builtin_shufflevector(lo, hb, 0,1,2,3,4,5,6,7,8,9,10,11,12,13,14,15);

    #pragma unroll
    for (int nt = 0; nt < 4; ++nt) {
      v16h bf = *(const v16h*)(&ldsB[buf][(nt * 16 + l15) * 32 + kbB]);
      acc[nt] = __builtin_amdgcn_wmma_f32_16x16x32_f16(false, a, false, bf,
                                                       (short)0, acc[nt], false, false);
    }

    wait_async0();
    __syncthreads();
  }

  #pragma unroll
  for (int r = 0; r < 8; ++r) {
    int mr = mtile * 16 + r + hi * 8;
    if (mr >= M) continue;
    #pragma unroll
    for (int nt = 0; nt < 4; ++nt) {
      int n = nbase + nt * 16 + l15;
      if (n >= N) continue;
      float v = acc[nt][r] + bias[n];
      if (act == 1)      v = v > 0.f ? v : 0.f;
      else if (act == 2) v = 1.f / (1.f + __expf(-v));
      if (out_f32) ((float*)outp)[(size_t)mr * ldc + n] = v;
      else         ((_Float16*)outp)[(size_t)mr * ldc + n] = (_Float16)v;
    }
  }
}

// ---------------------------------------------------------------- landmark gather: v[b][n][c]
__global__ void k_gather(const _Float16* __restrict__ xmid, const int* __restrict__ lm,
                         float* __restrict__ v, int Bn) {
  int i = blockIdx.x * blockDim.x + threadIdx.x;          // Bn*8*256
  if (i >= Bn * 2048) return;
  int c = i & 255; int n = (i >> 8) & 7; int b = i >> 11;
  const int* L = lm + (b * 8 + n) * 3;
  int vis = (L[0] == 1);
  int xi  = (int)((float)L[1] / 224.0f * 6.0f);
  int yi  = (int)((float)L[2] / 224.0f * 6.0f);
  int idx = xi * 6 + yi;
  float val = 0.f;
  if (vis) val = (float)xmid[((size_t)b * 256 + c) * 36 + idx];
  v[i] = val;
}

__global__ void k_gcn1(const float* __restrict__ v, const float* __restrict__ adj,
                       const float* __restrict__ w, const float* __restrict__ bias,
                       float* __restrict__ g1, int Bn) {
  int i = blockIdx.x * blockDim.x + threadIdx.x;          // Bn*8*16
  if (i >= Bn * 128) return;
  int f = i & 15; int n = (i >> 4) & 7; int b = i >> 7;
  float s = 0.f;
  for (int m = 0; m < 8; ++m) {
    float a = adj[n * 8 + m];
    const float* vb = v + (b * 8 + m) * 256;
    float d = 0.f;
    for (int c = 0; c < 256; ++c) d += vb[c] * w[c * 16 + f];
    s += a * d;
  }
  s += bias[f];
  g1[i] = s > 0.f ? s : 0.f;
}

__global__ void k_gcn2(const float* __restrict__ g1, const float* __restrict__ adj,
                       const float* __restrict__ w, const float* __restrict__ bias,
                       _Float16* __restrict__ xc, int Bn) {
  int i = blockIdx.x * blockDim.x + threadIdx.x;          // Bn*8*256
  if (i >= Bn * 2048) return;
  int f = i & 255; int n = (i >> 8) & 7; int b = i >> 11;
  float s = 0.f;
  for (int m = 0; m < 8; ++m) {
    float a = adj[n * 8 + m];
    const float* gb = g1 + (b * 8 + m) * 16;
    float d = 0.f;
    for (int c = 0; c < 16; ++c) d += gb[c] * w[c * 256 + f];
    s += a * d;
  }
  s += bias[f];
  xc[(size_t)b * 4096 + n * 256 + f] = (_Float16)s;
}

// ---------------------------------------------------------------- host
extern "C" void kernel_launch(void* const* d_in, const int* in_sizes, int n_in,
                              void* d_out, int out_size, void* d_ws, size_t ws_size,
                              hipStream_t stream) {
  (void)in_sizes; (void)n_in; (void)out_size; (void)ws_size;
  const int B = 256;

  const float* x     = (const float*)d_in[0];
  const int*   lm    = (const int*)d_in[1];
  const float* adj   = (const float*)d_in[2];
  const float* c1w   = (const float*)d_in[3];  const float* c1b  = (const float*)d_in[4];
  const float* c2w   = (const float*)d_in[5];  const float* c2b  = (const float*)d_in[6];
  const float* c3w   = (const float*)d_in[7];  const float* c3b  = (const float*)d_in[8];
  const float* c4w   = (const float*)d_in[9];  const float* c4b  = (const float*)d_in[10];
  const float* c5w   = (const float*)d_in[11]; const float* c5b  = (const float*)d_in[12];
  const float* gc1w  = (const float*)d_in[13]; const float* gc1b = (const float*)d_in[14];
  const float* gc2w  = (const float*)d_in[15]; const float* gc2b = (const float*)d_in[16];
  const float* fc1w  = (const float*)d_in[17]; const float* fc1b = (const float*)d_in[18];
  const float* fc2w  = (const float*)d_in[19]; const float* fc2b = (const float*)d_in[20];
  const float* cat1w = (const float*)d_in[21]; const float* cat1b = (const float*)d_in[22];
  const float* cat2w = (const float*)d_in[23]; const float* cat2b = (const float*)d_in[24];
  const float* att1w = (const float*)d_in[25]; const float* att1b = (const float*)d_in[26];
  const float* att2w = (const float*)d_in[27]; const float* att2b = (const float*)d_in[28];

  char* ws = (char*)d_ws;
  size_t cur = 0;
  auto alloc = [&](size_t elems, size_t esz) -> void* {
    void* p = ws + cur;
    cur += (elems * esz + 255) & ~(size_t)255;
    return p;
  };
  auto cdiv = [](long a, long b) -> int { return (int)((a + b - 1) / b); };

  _Float16* xh   = (_Float16*)alloc(38535168, 2);
  _Float16* w1   = (_Float16*)alloc((size_t)64 * 384, 2);         // conv1, Kpad=384
  _Float16* w2   = (_Float16*)alloc((size_t)192 * 1600, 2);
  _Float16* w3   = (_Float16*)alloc((size_t)384 * 1728, 2);
  _Float16* w4   = (_Float16*)alloc((size_t)256 * 3456, 2);
  _Float16* w5   = (_Float16*)alloc((size_t)256 * 2304, 2);
  _Float16* wfc1 = (_Float16*)alloc((size_t)4096 * 9216, 2);
  _Float16* wfc2 = (_Float16*)alloc((size_t)2048 * 4096, 2);
  _Float16* wct1 = (_Float16*)alloc((size_t)4096 * 4096, 2);
  _Float16* wct2 = (_Float16*)alloc((size_t)50 * 4096, 2);
  _Float16* wat1 = (_Float16*)alloc((size_t)4096 * 4096, 2);
  _Float16* wat2 = (_Float16*)alloc((size_t)1000 * 4096, 2);
  _Float16* bigA = (_Float16*)alloc(49561600, 2);                 // reused conv activations
  _Float16* p1   = (_Float16*)alloc(11943936, 2);                 // 256x64x27x27
  _Float16* p2   = (_Float16*)alloc(8306688, 2);                  // 256x192x13x13
  _Float16* xmid = (_Float16*)alloc(2359296, 2);                  // 256x256x6x6
  _Float16* hfc1 = (_Float16*)alloc(1048576, 2);
  _Float16* xc   = (_Float16*)alloc(1048576, 2);
  _Float16* hcat = (_Float16*)alloc(1048576, 2);
  _Float16* hatt = (_Float16*)alloc(1048576, 2);
  float*    vbuf = (float*)alloc(524288, 4);
  float*    g1   = (float*)alloc(32768, 4);

  // ---- convert inputs / weights to f16
  k_f32_to_f16<<<cdiv(38535168, 256), 256, 0, stream>>>(x, xh, 38535168);
  k_w_pad_f16<<<cdiv(64 * 384, 256), 256, 0, stream>>>(c1w, w1, 64, 363, 384);
  k_f32_to_f16<<<cdiv(307200, 256), 256, 0, stream>>>(c2w, w2, 307200);
  k_f32_to_f16<<<cdiv(663552, 256), 256, 0, stream>>>(c3w, w3, 663552);
  k_f32_to_f16<<<cdiv(884736, 256), 256, 0, stream>>>(c4w, w4, 884736);
  k_f32_to_f16<<<cdiv(589824, 256), 256, 0, stream>>>(c5w, w5, 589824);
  k_f32_to_f16<<<cdiv(37748736, 256), 256, 0, stream>>>(fc1w, wfc1, 37748736);
  k_f32_to_f16<<<cdiv(8388608, 256), 256, 0, stream>>>(fc2w, wfc2, 8388608);
  k_f32_to_f16<<<cdiv(16777216, 256), 256, 0, stream>>>(cat1w, wct1, 16777216);
  k_f32_to_f16<<<cdiv(204800, 256), 256, 0, stream>>>(cat2w, wct2, 204800);
  k_f32_to_f16<<<cdiv(16777216, 256), 256, 0, stream>>>(att1w, wat1, 16777216);
  k_f32_to_f16<<<cdiv(4096000, 256), 256, 0, stream>>>(att2w, wat2, 4096000);

  // ---- conv stack (implicit-GEMM WMMA, async-LDS weight staging, bias+relu fused)
  k_conv_wmma<11,11,4,2><<<dim3(48400 / 8, 1), 256, 0, stream>>>(xh, w1, c1b, bigA,
      B, 3, 224, 224, 64, 55, 55, 363, 384);
  k_maxpool3s2<<<cdiv((long)B * 64 * 27 * 27, 256), 256, 0, stream>>>(
      bigA, p1, 64, 55, 55, 27, 27, B * 64 * 27 * 27);
  k_conv_wmma<5,5,1,2><<<dim3(11664 / 8, 3), 256, 0, stream>>>(p1, w2, c2b, bigA,
      B, 64, 27, 27, 192, 27, 27, 1600, 1600);
  k_maxpool3s2<<<cdiv((long)B * 192 * 13 * 13, 256), 256, 0, stream>>>(
      bigA, p2, 192, 27, 27, 13, 13, B * 192 * 13 * 13);
  k_conv_wmma<3,3,1,1><<<dim3(2704 / 8, 6), 256, 0, stream>>>(p2, w3, c3b, bigA,
      B, 192, 13, 13, 384, 13, 13, 1728, 1728);
  _Float16* a4 = bigA + 16605696;
  k_conv_wmma<3,3,1,1><<<dim3(2704 / 8, 4), 256, 0, stream>>>(bigA, w4, c4b, a4,
      B, 384, 13, 13, 256, 13, 13, 3456, 3456);
  k_conv_wmma<3,3,1,1><<<dim3(2704 / 8, 4), 256, 0, stream>>>(a4, w5, c5b, bigA,
      B, 256, 13, 13, 256, 13, 13, 2304, 2304);
  k_maxpool3s2<<<cdiv((long)B * 256 * 36, 256), 256, 0, stream>>>(
      bigA, xmid, 256, 13, 13, 6, 6, B * 256 * 36);

  // ---- landmark GCN branch (tiny, fp32) -> xc columns [0,2048)
  k_gather<<<cdiv((long)B * 2048, 256), 256, 0, stream>>>(xmid, lm, vbuf, B);
  k_gcn1<<<cdiv((long)B * 128, 256), 256, 0, stream>>>(vbuf, adj, gc1w, gc1b, g1, B);
  k_gcn2<<<cdiv((long)B * 2048, 256), 256, 0, stream>>>(g1, adj, gc2w, gc2b, xc, B);

  // ---- FC branch (WMMA GEMMs). fc2 writes xc columns [2048,4096) via ldc offset.
  k_gemm_wmma<<<dim3(2, 64), 256, 0, stream>>>(xmid, wfc1, fc1b, hfc1, 256, 4096, 9216, 4096, 1, 0);
  k_gemm_wmma<<<dim3(2, 32), 256, 0, stream>>>(hfc1, wfc2, fc2b, xc + 2048, 256, 2048, 4096, 4096, 0, 0);

  // ---- heads: category -> d_out[0:12800), attribute (sigmoid) -> d_out[12800:...)
  k_gemm_wmma<<<dim3(2, 64), 256, 0, stream>>>(xc, wct1, cat1b, hcat, 256, 4096, 4096, 4096, 1, 0);
  k_gemm_wmma<<<dim3(2, 1), 256, 0, stream>>>(hcat, wct2, cat2b, d_out, 256, 50, 4096, 50, 0, 1);
  k_gemm_wmma<<<dim3(2, 64), 256, 0, stream>>>(xc, wat1, att1b, hatt, 256, 4096, 4096, 4096, 1, 0);
  k_gemm_wmma<<<dim3(2, 16), 256, 0, stream>>>(hatt, wat2, att2b, (float*)d_out + 256 * 50,
                                               256, 1000, 4096, 1000, 2, 1);
}
